// SelfAttention_74002286510407
// MI455X (gfx1250) — matrix operand order repaired and checked by reading, not compiled, not run on hardware
//
#include <hip/hip_runtime.h>

// ---------------------------------------------------------------------------
// Self-attention (B=8192, C=4096, H=32, hd=128) for MI455X / gfx1250.
// GEMM1 (x @ w_qkv^T, f32->bf16 on the fly, WMMA) -> per-token 32x32
// attention over heads (WMMA, async-LDS staged V) -> GEMM2 (attn @ w_proj^T
// + bias, async-LDS staged bf16 A, WMMA, f32 out).
// Workspace: qkv bf16 (192MB) + permuted attn out bf16 (64MB) = 256MB.
// ---------------------------------------------------------------------------

typedef __attribute__((ext_vector_type(16))) __bf16          bf16x16;
typedef __attribute__((ext_vector_type(8)))  float           f32x8;
typedef __attribute__((ext_vector_type(8)))  unsigned short  u16x8;
typedef __attribute__((ext_vector_type(16))) unsigned short  u16x16;
typedef __attribute__((ext_vector_type(4)))  unsigned int    u32x4;

// GCC-vector int4 to match the async-LDS builtin's parameter type exactly
typedef int i32x4 __attribute__((vector_size(16)));
typedef __attribute__((address_space(1))) i32x4* gas_i32x4;   // global int4*
typedef __attribute__((address_space(3))) i32x4* las_i32x4;   // LDS int4*

#if defined(__has_builtin)
#  if __has_builtin(__builtin_amdgcn_global_load_async_to_lds_b128)
#    define HAVE_ASYNC_LDS 1
#  endif
#  if __has_builtin(__builtin_amdgcn_s_wait_asynccnt)
#    define HAVE_WAIT_ASYNC 1
#  endif
#  if __has_builtin(__builtin_amdgcn_cvt_pk_bf16_f32)
#    define HAVE_CVT_PK 1
#  endif
#endif

union FragU { u16x16 u; bf16x16 b; };

__device__ __forceinline__ unsigned short f2bf(float f) {
  unsigned int u = __float_as_uint(f);
  u += 0x7FFFu + ((u >> 16) & 1u);      // round to nearest even
  return (unsigned short)(u >> 16);
}

// pack two f32 -> one dword of 2x bf16 (hw packed cvt when available)
__device__ __forceinline__ unsigned int pkbf(float lo, float hi) {
#ifdef HAVE_CVT_PK
  typedef __attribute__((ext_vector_type(2))) __bf16 bf16x2;
  union { bf16x2 v; unsigned int u; } r;
  r.v = __builtin_amdgcn_cvt_pk_bf16_f32(lo, hi);
  return r.u;
#else
  return (unsigned int)f2bf(lo) | ((unsigned int)f2bf(hi) << 16);
#endif
}

// 16-byte global -> LDS async copy (ASYNCcnt tracked); plain copy fallback
__device__ __forceinline__ void async_cp16(const unsigned short* g, unsigned short* l) {
#ifdef HAVE_ASYNC_LDS
  // global ptr: flat address == AS1 address, cast via integer
  gas_i32x4 gp = (gas_i32x4)(unsigned long long)(const void*)g;
  // LDS ptr: flat-LDS addr[31:0] is the LDS offset (ISA 10.2 aperture table)
  las_i32x4 lp = (las_i32x4)(unsigned int)(unsigned long long)(const void*)l;
  __builtin_amdgcn_global_load_async_to_lds_b128(gp, lp, 0, 0);
#else
  *(u16x8*)l = *(const u16x8*)g;
#endif
}

__device__ __forceinline__ void async_wait0() {
#ifdef HAVE_ASYNC_LDS
# ifdef HAVE_WAIT_ASYNC
  __builtin_amdgcn_s_wait_asynccnt(0);
# else
  asm volatile("s_wait_asynccnt 0" ::: "memory");
# endif
#endif
}

__device__ __forceinline__ bf16x16 mkfrag(u16x8 lo, u16x8 hi) {
  FragU r;
  r.u = __builtin_shufflevector(lo, hi, 0,1,2,3,4,5,6,7,8,9,10,11,12,13,14,15);
  return r.b;
}

__device__ __forceinline__ f32x8 wmma_bf16(bf16x16 a, bf16x16 b, f32x8 c) {
  return __builtin_amdgcn_wmma_f32_16x16x32_bf16(false, a, false, b,
                                                 (short)0, c, false, false);
}

__device__ __forceinline__ void load32f(const float* __restrict__ src, float* f) {
  #pragma unroll
  for (int i = 0; i < 32; ++i) f[i] = src[i];
}

__device__ __forceinline__ void pack_store32(unsigned short* dst, const float* f) {
  #pragma unroll
  for (int i = 0; i < 4; ++i) {
    u32x4 v;
    #pragma unroll
    for (int j = 0; j < 4; ++j) v[j] = pkbf(f[8 * i + 2 * j], f[8 * i + 2 * j + 1]);
    ((u32x4*)dst)[i] = v;
  }
}

// ---------------------------------------------------------------------------
// NT GEMM: C[M,N] = A[M,K] * B[N,K]^T. 256 thr = 8 waves (4 M x 2 N),
// WG tile 128x128, K step 64, double-buffered LDS (bf16), wave tile 32x64.
// A path: f32 (VGPR-staged + packed cvt) or bf16 (async DMA to LDS).
// ---------------------------------------------------------------------------
template<bool A_BF16, bool OUT_F32>
__global__ __launch_bounds__(256)
void gemm_nt(const void* __restrict__ Aptr, const float* __restrict__ Bw,
             const float* __restrict__ bias, void* __restrict__ Cptr,
             int M, int N, int K)
{
  __shared__ __align__(16) unsigned short As[2][128][64];   // 32 KB
  __shared__ __align__(16) unsigned short Bs[2][128][64];   // 32 KB

  const int t      = threadIdx.x;
  const int lane   = t & 31;
  const int wave   = t >> 5;
  const int waveM  = wave & 3;
  const int waveN  = wave >> 2;
  const int mBase  = blockIdx.y * 128;
  const int nBase  = blockIdx.x * 128;

  const int srow   = t >> 1;          // staging row 0..127
  const int scol   = (t & 1) * 32;    // staging col 0 / 32

  const int frow   = lane & 15;
  const int hiHalf = lane >> 4;

  const float*          Af = (const float*)Aptr;
  const unsigned short* Ab = (const unsigned short*)Aptr;

  const size_t aRowOff = (size_t)(mBase + srow) * (size_t)K + (size_t)scol;
  const size_t bRowOff = (size_t)(nBase + srow) * (size_t)K + (size_t)scol;

  f32x8 zero = {};
  f32x8 acc[2][4];
  #pragma unroll
  for (int m = 0; m < 2; ++m)
    #pragma unroll
    for (int n = 0; n < 4; ++n) acc[m][n] = zero;

  const int NT = K >> 6;
  float fa[32], fb[32];

  // ---- prologue: stage tile 0 into buffer 0 ----
  if (A_BF16) {
    #pragma unroll
    for (int i = 0; i < 4; ++i)
      async_cp16(Ab + aRowOff + i * 8, &As[0][srow][scol] + i * 8);
  } else {
    load32f(Af + aRowOff, fa);
    pack_store32(&As[0][srow][scol], fa);
  }
  load32f(Bw + bRowOff, fb);
  pack_store32(&Bs[0][srow][scol], fb);
  async_wait0();
  __syncthreads();

  for (int kt = 0; kt < NT; ++kt) {
    const int  cur   = kt & 1;
    const int  nxt   = cur ^ 1;
    const bool more  = (kt + 1) < NT;
    const int  knext = (kt + 1) << 6;

    // issue next-tile traffic before compute so it overlaps the WMMAs
    if (more) {
      if (A_BF16) {
        #pragma unroll
        for (int i = 0; i < 4; ++i)
          async_cp16(Ab + aRowOff + knext + i * 8, &As[nxt][srow][scol] + i * 8);
      } else {
        load32f(Af + aRowOff + knext, fa);
      }
      load32f(Bw + bRowOff + knext, fb);
    }
    if (kt + 2 < NT) {
      __builtin_prefetch(Bw + bRowOff + ((kt + 2) << 6), 0, 0);
      if (!A_BF16) __builtin_prefetch(Af + aRowOff + ((kt + 2) << 6), 0, 0);
    }

    // ---- compute on buffer `cur`: 2 k-substeps x (2 M x 4 N) WMMAs ----
    #pragma unroll
    for (int kk = 0; kk < 64; kk += 32) {
      bf16x16 afrag[2];
      #pragma unroll
      for (int m = 0; m < 2; ++m) {
        const unsigned short* p = &As[cur][waveM * 32 + m * 16 + frow][kk + hiHalf * 16];
        afrag[m] = mkfrag(*(const u16x8*)p, *(const u16x8*)(p + 8));
      }
      #pragma unroll
      for (int n = 0; n < 4; ++n) {
        const unsigned short* p = &Bs[cur][waveN * 64 + n * 16 + frow][kk + hiHalf * 16];
        bf16x16 bfrag = mkfrag(*(const u16x8*)p, *(const u16x8*)(p + 8));
        #pragma unroll
        for (int m = 0; m < 2; ++m)
          acc[m][n] = wmma_bf16(afrag[m], bfrag, acc[m][n]);
      }
    }

    if (more) {
      if (!A_BF16) pack_store32(&As[nxt][srow][scol], fa);
      pack_store32(&Bs[nxt][srow][scol], fb);
    }
    async_wait0();
    __syncthreads();
  }

  // ---- epilogue ----
  #pragma unroll
  for (int m = 0; m < 2; ++m) {
    const int row0 = mBase + waveM * 32 + m * 16 + hiHalf * 8;
    #pragma unroll
    for (int n = 0; n < 4; ++n) {
      const int col = nBase + waveN * 64 + n * 16 + frow;
      #pragma unroll
      for (int r = 0; r < 8; ++r) {
        size_t off = (size_t)(row0 + r) * (size_t)N + (size_t)col;
        if (OUT_F32) {
          ((float*)Cptr)[off] = acc[m][n][r] + bias[col];
        } else {
          ((unsigned short*)Cptr)[off] = f2bf(acc[m][n][r]);
        }
      }
    }
  }
}

// ---------------------------------------------------------------------------
// Per-token attention: one wave per token. qkv row b (bf16, 12288):
// q=[0,4096) k=[4096,8192) v=[8192,12288), each 32 rows x 128.
// V is async-DMA'd to LDS at kernel start (overlaps S=q k^T WMMAs).
// ---------------------------------------------------------------------------
__global__ __launch_bounds__(256)
void attn_kernel(const unsigned short* __restrict__ qkv,
                 unsigned short* __restrict__ out)
{
  __shared__ __align__(16) float          Sbuf[8][32][32];   // 32 KB
  __shared__ __align__(16) unsigned short Pbuf[8][32][32];   // 16 KB
  __shared__ __align__(16) unsigned short Vbuf[8][32][128];  // 64 KB

  const int lane   = threadIdx.x & 31;
  const int wave   = threadIdx.x >> 5;
  const long b     = (long)blockIdx.x * 8 + wave;
  const unsigned short* qp = qkv + b * 12288;
  const unsigned short* kp = qp + 4096;
  const unsigned short* vp = qp + 8192;

  const int frow   = lane & 15;
  const int hiHalf = lane >> 4;

  // ---- kick off async copy of V (32x128 bf16 = 8KB) into LDS ----
  {
    unsigned short* vl = &Vbuf[wave][0][0];
    #pragma unroll
    for (int i = 0; i < 16; ++i) {
      const int e = (i * 32 + lane) * 8;   // 16B chunks
      async_cp16(vp + e, vl + e);
    }
  }

  f32x8 zero = {};
  f32x8 s[2][2];
  #pragma unroll
  for (int m = 0; m < 2; ++m)
    #pragma unroll
    for (int n = 0; n < 2; ++n) s[m][n] = zero;

  // ---- S = q k^T (WMMA straight from global) ----
  #pragma unroll
  for (int d0 = 0; d0 < 128; d0 += 32) {
    bf16x16 aq[2];
    #pragma unroll
    for (int m = 0; m < 2; ++m) {
      const unsigned short* p = qp + (m * 16 + frow) * 128 + d0 + hiHalf * 16;
      aq[m] = mkfrag(*(const u16x8*)p, *(const u16x8*)(p + 8));
    }
    #pragma unroll
    for (int n = 0; n < 2; ++n) {
      const unsigned short* p = kp + (n * 16 + frow) * 128 + d0 + hiHalf * 16;
      bf16x16 bk = mkfrag(*(const u16x8*)p, *(const u16x8*)(p + 8));
      #pragma unroll
      for (int m = 0; m < 2; ++m)
        s[m][n] = wmma_bf16(aq[m], bk, s[m][n]);
    }
  }

  // ---- spill S ----
  #pragma unroll
  for (int m = 0; m < 2; ++m)
    #pragma unroll
    for (int n = 0; n < 2; ++n)
      #pragma unroll
      for (int r = 0; r < 8; ++r)
        Sbuf[wave][m * 16 + hiHalf * 8 + r][n * 16 + frow] = s[m][n][r];
  __syncthreads();

  // ---- softmax: lane owns row h = lane ----
  {
    const float scale = 0.08838834764831845f;   // 128^-0.5
    const float* row = &Sbuf[wave][lane][0];
    float mx = -3.0e38f;
    #pragma unroll
    for (int g = 0; g < 32; ++g) mx = fmaxf(mx, row[g]);
    float ex[32];
    float sum = 0.0f;
    #pragma unroll
    for (int g = 0; g < 32; ++g) { ex[g] = __expf(scale * (row[g] - mx)); sum += ex[g]; }
    const float inv = 1.0f / sum;
    #pragma unroll
    for (int g = 0; g < 32; ++g) Pbuf[wave][lane][g] = f2bf(ex[g] * inv);
  }
  __syncthreads();

  // ---- O = P (32x32) x V (32x128) ----
  bf16x16 ap[2];
  #pragma unroll
  for (int m = 0; m < 2; ++m) {
    const unsigned short* p = &Pbuf[wave][m * 16 + frow][hiHalf * 16];
    ap[m] = mkfrag(*(const u16x8*)p, *(const u16x8*)(p + 8));
  }
  async_wait0();   // V resident in LDS
  #pragma unroll
  for (int n = 0; n < 8; ++n) {
    const int d = n * 16 + frow;
    FragU bv;
    #pragma unroll
    for (int e = 0; e < 16; ++e)
      bv.u[e] = Vbuf[wave][hiHalf * 16 + e][d];   // v[g][d] column gather (LDS)
    #pragma unroll
    for (int m = 0; m < 2; ++m) {
      f32x8 o = wmma_bf16(ap[m], bv.b, zero);
      // store permuted: out[b*4096 + d*32 + h], h contiguous over r
      u16x8 pk;
      #pragma unroll
      for (int r = 0; r < 8; ++r) pk[r] = f2bf(o[r]);
      *(u16x8*)(out + b * 4096 + (long)d * 32 + m * 16 + hiHalf * 8) = pk;
    }
  }
}

// ---------------------------------------------------------------------------
extern "C" void kernel_launch(void* const* d_in, const int* in_sizes, int n_in,
                              void* d_out, int out_size, void* d_ws, size_t ws_size,
                              hipStream_t stream) {
  const float* x      = (const float*)d_in[0];   // (8192, 4096)
  const float* w_qkv  = (const float*)d_in[1];   // (12288, 4096)
  const float* w_proj = (const float*)d_in[2];   // (4096, 4096)
  const float* b_proj = (const float*)d_in[3];   // (4096,)
  float*       outp   = (float*)d_out;           // (8192, 4096)

  const int M = 8192, C = 4096, N1 = 12288;

  unsigned short* qkv_ws  = (unsigned short*)d_ws;              // 192 MB bf16
  unsigned short* attn_ws = qkv_ws + (size_t)M * (size_t)N1;    //  64 MB bf16

  dim3 g1(N1 / 128, M / 128);
  gemm_nt<false, false><<<g1, 256, 0, stream>>>(x, w_qkv, nullptr, qkv_ws, M, N1, C);

  attn_kernel<<<M / 8, 256, 0, stream>>>(qkv_ws, attn_ws);

  dim3 g2(C / 128, M / 128);
  gemm_nt<true, true><<<g2, 256, 0, stream>>>(attn_ws, w_proj, b_proj, outp, M, C, C);
}
